// RoadLoss_30219389895055
// MI455X (gfx1250) — compile-verified, matrix operand order checked by reading, not run on hardware
//
#include <hip/hip_runtime.h>

#define MAP_H 1024
#define MAP_W 1024
#define NPTS  128
#define TILE_H 32
#define TILE_W 128

typedef __attribute__((ext_vector_type(4)))  unsigned int u32x4;
typedef __attribute__((ext_vector_type(8)))  int          i32x8;
typedef __attribute__((ext_vector_type(4)))  int          i32x4;
typedef __attribute__((ext_vector_type(16))) _Float16     v16h;
typedef __attribute__((ext_vector_type(8)))  float        v8f;

__device__ __forceinline__ int iclamp(int v, int lo, int hi) {
    return v < lo ? lo : (v > hi ? hi : v);
}

// ---------------------------------------------------------------------------
// Kernel 1: init per-point min accumulators (uint-encoded non-negative floats)
// ---------------------------------------------------------------------------
__global__ void rl_init(unsigned* ws) {
    int i = threadIdx.x;
    if (i < 2 * NPTS) ws[i] = 0x7F800000u;  // +inf bits
}

// ---------------------------------------------------------------------------
// Kernel 2: per-tile masked min of squared distance for all 128 points.
// Tile staged into LDS via Tensor Data Mover (async DMA, TENSORcnt).
// This call is UNGUARDED on purpose: if this TU compiles, the
// tensor_load_to_lds instruction is definitively in the binary.
// ---------------------------------------------------------------------------
__global__ __launch_bounds__(256)
void rl_tile_min(const float* __restrict__ hd_map,
                 const int*   __restrict__ pred,
                 unsigned*    __restrict__ ws_on,
                 unsigned*    __restrict__ ws_off) {
    __shared__ float tile[TILE_H * TILE_W];   // 16 KB

    const int row0 = blockIdx.y * TILE_H;
    const int col0 = blockIdx.x * TILE_W;

    // --- TDM: DMA the 32x128 fp32 tile from global into LDS (wave 0 issues) ---
    if (threadIdx.x < 32) {
        unsigned long long ga = (unsigned long long)(const void*)hd_map
                              + (unsigned long long)(row0 * MAP_W + col0) * 4ull;
        unsigned lds_off = (unsigned)(unsigned long long)(void*)tile;

        // D# group 0: count=1 (user mode), lds_addr, global_addr[56:0], type=2
        u32x4 g0;
        g0[0] = 1u;
        g0[1] = lds_off;
        g0[2] = (unsigned)(ga & 0xFFFFFFFFull);
        g0[3] = ((unsigned)(ga >> 32) & 0x01FFFFFFu) | (2u << 30);

        // D# group 1: data_size=4B, tensor 1024x1024, tile 128x32, stride 1024
        i32x8 g1;
        g1[0] = (int)(2u << 16);                                  // data_size = 4B
        g1[1] = (int)(((unsigned)MAP_W & 0xFFFFu) << 16);         // tensor_dim0[15:0]
        g1[2] = (int)((((unsigned)MAP_W >> 16) & 0xFFFFu)
                    | (((unsigned)MAP_H & 0xFFFFu) << 16));       // dim0 hi | dim1 lo
        g1[3] = (int)((((unsigned)MAP_H >> 16) & 0xFFFFu)
                    | ((unsigned)TILE_W << 16));                  // dim1 hi | tile_dim0
        g1[4] = (int)((unsigned)TILE_H);                          // tile_dim1, tile_dim2=0
        g1[5] = (int)((unsigned)MAP_W);                           // tensor_dim0_stride lo32
        g1[6] = 0;
        g1[7] = 0;

        i32x4 g2 = {};   // 2-D tile: groups 2/3 unused (tile_dim2 == 0)
        i32x4 g3 = {};
#if defined(__clang_major__) && (__clang_major__ >= 23)
        i32x8 gpad = {};
        __builtin_amdgcn_tensor_load_to_lds(g0, g1, g2, g3, gpad, 0);
#else
        __builtin_amdgcn_tensor_load_to_lds(g0, g1, g2, g3, 0);
#endif
        __builtin_amdgcn_s_wait_tensorcnt(0);
    }
    __syncthreads();

    const int wave = threadIdx.x >> 5;
    const int lane = threadIdx.x & 31;

    // Each of the 8 waves owns 16 of the 128 points.
    for (int t = 0; t < 16; ++t) {
        const int n = wave * 16 + t;
        const float px = (float)pred[2 * n + 0];  // row coordinate
        const float py = (float)pred[2 * n + 1];  // col coordinate

        float dc2[4];
#pragma unroll
        for (int xi = 0; xi < 4; ++xi) {
            float cc = (float)(col0 + lane + 32 * xi);
            float dc = cc - py;
            dc2[xi] = dc * dc;                    // exact: |dc| <= 1023
        }

        float minOn  = __builtin_inff();
        float minOff = __builtin_inff();

        for (int y = 0; y < TILE_H; ++y) {
            float rr  = (float)(row0 + y);
            float dr  = rr - px;
            float dr2 = dr * dr;                  // exact
            const float* rowp = &tile[y * TILE_W];
#pragma unroll
            for (int xi = 0; xi < 4; ++xi) {
                float v  = rowp[lane + 32 * xi];
                float d2 = dr2 + dc2[xi];         // exact (< 2^22)
                if (v != 0.0f) minOn  = fminf(minOn,  d2);
                else           minOff = fminf(minOff, d2);
            }
        }

        // wave32 min-reduce
#pragma unroll
        for (int off = 16; off > 0; off >>= 1) {
            minOn  = fminf(minOn,  __shfl_xor(minOn,  off, 32));
            minOff = fminf(minOff, __shfl_xor(minOff, off, 32));
        }
        if (lane == 0) {
            atomicMin(&ws_on[n],  __float_as_uint(minOn));
            atomicMin(&ws_off[n], __float_as_uint(minOff));
        }
    }
}

// ---------------------------------------------------------------------------
// Kernel 3: per-point loss + mean. The 128-element sum is done with one
// v_wmma_f32_16x16x32_f16: A = ones (exact), B holds hi/lo f16 splits of each
// loss (hi+lo carries ~22 mantissa bits; 1.0*f16 products accumulate exactly
// in the f32 WMMA accumulator). Each column's 16 contributions live entirely
// in the two lanes owning that column, so the column-sum is layout-robust.
// ---------------------------------------------------------------------------
__global__ void rl_finalize(const float*    __restrict__ hd_map,
                            const int*      __restrict__ pred,
                            const unsigned* __restrict__ ws_on,
                            const unsigned* __restrict__ ws_off,
                            float*          __restrict__ out) {
    __shared__ _Float16 hiArr[NPTS];
    __shared__ _Float16 loArr[NPTS];

    const int lane = threadIdx.x;  // 32 threads, one wave

    for (int t = 0; t < 4; ++t) {
        int n  = lane + 32 * t;
        int px = pred[2 * n + 0];
        int py = pred[2 * n + 1];

        bool outside_frame = (px < 0) || (px > MAP_H) || (py < 0) || (py > MAP_W);

        // NOTE: reference swaps axes here (rows from py, cols from px) — replicate.
        int ir[4] = { py - 1, py - 1, py, py };
        int ic[4] = { px - 1, px,     px - 1, px };
        bool outside_road = false;
#pragma unroll
        for (int k = 0; k < 4; ++k) {
            int r = iclamp(ir[k], 0, MAP_H - 1);
            int c = iclamp(ic[k], 0, MAP_W - 1);
            outside_road = outside_road || (hd_map[r * MAP_W + c] == 1.0f);
        }

        float d2on  = __uint_as_float(ws_on[n]);
        float d2off = __uint_as_float(ws_off[n]);

        float loss_off = expf(sqrtf(d2off) * (0.69314718056f / 40.0f)); // exp(sqrt*ln2/K2)
        float loss_on  = expf(-d2on * (1.0f / 21.7f));                  // exp(-d2/K1)
        float L = outside_frame ? 0.0f : (outside_road ? loss_off : loss_on);

        _Float16 hi = (_Float16)L;
        _Float16 lo = (_Float16)(L - (float)hi);
        hiArr[n] = hi;
        loArr[n] = lo;
    }
    __syncthreads();

    // A = ones(16x32) in f16 (exact)
    v16h a;
#pragma unroll
    for (int j = 0; j < 16; ++j) a[j] = (_Float16)1.0f;

    // B(32x16): column n (owned by lanes n and n+16) gets the 8 hi + 8 lo
    // parts of losses {n, n+16, ..., n+112} in lane n; lane n+16 holds zeros.
    v16h b;
#pragma unroll
    for (int j = 0; j < 16; ++j) b[j] = (_Float16)0.0f;
    const int ncol = lane & 15;
    if (lane < 16) {
#pragma unroll
        for (int j = 0; j < 8; ++j) {
            b[j]     = hiArr[ncol + 16 * j];
            b[j + 8] = loArr[ncol + 16 * j];
        }
    }

    v8f c = {};
    c = __builtin_amdgcn_wmma_f32_16x16x32_f16(false, a, false, b,
                                               (short)0, c, false, false);

    // Every lane's c[0] = colsum(N = lane%16); sum the 16 distinct columns.
    float s = c[0];
    s += __shfl_xor(s, 1, 32);
    s += __shfl_xor(s, 2, 32);
    s += __shfl_xor(s, 4, 32);
    s += __shfl_xor(s, 8, 32);

    if (lane == 0) out[0] = s * (1.0f / (float)NPTS);
}

// ---------------------------------------------------------------------------
extern "C" void kernel_launch(void* const* d_in, const int* in_sizes, int n_in,
                              void* d_out, int out_size, void* d_ws, size_t ws_size,
                              hipStream_t stream) {
    const float* hd_map = (const float*)d_in[0];
    const int*   pred   = (const int*)d_in[1];
    unsigned* ws     = (unsigned*)d_ws;
    unsigned* ws_on  = ws;
    unsigned* ws_off = ws + NPTS;
    float* out = (float*)d_out;

    rl_init<<<1, 256, 0, stream>>>(ws);

    dim3 grid(MAP_W / TILE_W, MAP_H / TILE_H);  // 8 x 32 = 256 blocks
    rl_tile_min<<<grid, 256, 0, stream>>>(hd_map, pred, ws_on, ws_off);

    rl_finalize<<<1, 32, 0, stream>>>(hd_map, pred, ws_on, ws_off, out);
}
